// Architecture_73297911874155
// MI455X (gfx1250) — matrix operand order repaired and checked by reading, not compiled, not run on hardware
//
#include <hip/hip_runtime.h>

typedef __attribute__((ext_vector_type(16))) _Float16 v16h;
typedef __attribute__((ext_vector_type(8)))  _Float16 v8h;
typedef __attribute__((ext_vector_type(8)))  float    v8f;

#define PX    1024           // 32x32 interior pixels
#define PDIM  34             // padded spatial dim (1-pixel zero halo)
#define PPX   (PDIM * PDIM)  // 1156 padded pixels
#define BATCH 32

__device__ __forceinline__ int padOff(int pix) {
    return ((pix >> 5) + 1) * PDIM + (pix & 31) + 1;
}

// ---------------------------------------------------------------------------
// Weight packing: f32 OIHW (or ConvTranspose (Cin,Cout,3,3) with flip) ->
// f16 WMMA A-fragment layout (16x32 tile, per-lane v16h, contiguous K order):
//   lane 0-15 : M=lane,    halves 0-15 -> K=0..15
//   lane 16-31: M=lane-16, halves 0-15 -> K=16..31
// Logical K is TAP-MAJOR: k = tap*Cin + ci (tap = kh*3+kw): each 16-wide K
// segment = one tap + 16 contiguous NHWC channels. k >= Cin*9 packs as ZERO,
// which lets the conv kernel load B unconditionally for the K tail.
// Tile order: [mt][kt][lane][half] so the K-loop strides 512 halves.
// ---------------------------------------------------------------------------
__global__ void pack_w_k(const float* __restrict__ W, _Float16* __restrict__ Wp,
                         int Cout, int Cin, int transpose, int total)
{
    int i = blockIdx.x * blockDim.x + threadIdx.x;
    if (i >= total) return;
    int KT   = (Cin * 9 + 31) >> 5;
    int h    = i & 15;
    int lane = (i >> 4) & 31;
    int tile = i >> 9;
    int kt   = tile % KT;
    int mt   = tile / KT;
    int m    = mt * 16 + (lane & 15);
    int klocal = h + ((lane >= 16) ? 16 : 0);
    int k    = kt * 32 + klocal;
    float v = 0.f;
    if (m < Cout && k < Cin * 9) {
        int tap = k / Cin, ci = k - tap * Cin;
        int kh = tap / 3, kw = tap - kh * 3;
        if (!transpose)
            v = W[(((size_t)m * Cin + ci) * 3 + kh) * 3 + kw];
        else  // ConvTranspose2d: w2[o][i][kh][kw] = w[i][o][2-kh][2-kw]
            v = W[(((size_t)ci * Cout + m) * 3 + (2 - kh)) * 3 + (2 - kw)];
    }
    Wp[i] = (_Float16)v;
}

// constexpr B-fragment byte-element offset for K segment `seg` (16 k values):
// tap-shifted pixel (halo-padded NHWC) + channel block. Segments past the end
// of K clamp to the last valid one (their A columns are zero-packed).
template <int CIN>
__host__ __device__ constexpr int bOff(int seg) {
    int nseg = 9 * (CIN / 16);
    int s = (seg < nseg) ? seg : nseg - 1;
    int tap = s / (CIN / 16);
    int ci0 = (s - tap * (CIN / 16)) * 16;
    int dy = tap / 3 - 1, dx = tap - (tap / 3) * 3 - 1;
    return (dy * PDIM + dx) * CIN + ci0;
}

// ---------------------------------------------------------------------------
// 3x3 pad-1 conv on 32x32 maps, halo-padded NHWC, implicit GEMM via WMMA.
//   M = Cout (16/tile), N = pixels (16/tile), K = Cin*9 (tap-major, 32/step)
// grid = (Cout/16, 8, BATCH), block = 256 (8 waves; wave w -> ntile=by*8+w)
// Inner loop per chunk: 1 unconditional v16h weight load, 1 unconditional
// v16h activation load at a constexpr offset, 1 v_wmma. No branches.
// Outputs: optional f32 dense NHWC (v8f store, beta accumulate) and optional
// f16 halo-padded NHWC concat buffer (Ctot channels, offset c0, v8h store).
// ---------------------------------------------------------------------------
template <int CIN>
__global__ void conv3x3_wmma(const _Float16* __restrict__ in,
                             const _Float16* __restrict__ Wp,
                             const float* __restrict__ bias, int Cout,
                             float* __restrict__ outF, int beta,
                             _Float16* __restrict__ outH, int CtotH, int c0H)
{
    constexpr int KT = (CIN * 9 + 31) / 32;

    const int lane = threadIdx.x & 31;
    const int wv   = threadIdx.x >> 5;
    const int mt   = blockIdx.x;
    const int nt   = blockIdx.y * 8 + wv;
    const int b    = blockIdx.z;

    const int nl  = lane & 15;
    const int pix = nt * 16 + nl;
    const int hi  = lane >> 4;

    v8f acc = {};
    const _Float16* wbase = Wp + (size_t)mt * KT * 512 + lane * 16;
    const _Float16* bcen  = in + ((size_t)b * PPX + padOff(pix)) * CIN;

#pragma unroll
    for (int kt = 0; kt < KT; ++kt) {
        v16h a = *(const v16h*)(wbase + kt * 512);
        const int o0 = bOff<CIN>(2 * kt);
        const int o1 = bOff<CIN>(2 * kt + 1);
        v16h bf = *(const v16h*)(bcen + (hi ? o1 : o0));
        acc = __builtin_amdgcn_wmma_f32_16x16x32_f16(
            false, a, false, bf, (short)0, acc, false, false);
    }

    const int co0 = mt * 16 + hi * 8;
    v8f o;
#pragma unroll
    for (int r = 0; r < 8; ++r) o[r] = acc[r] + bias[co0 + r];

    if (outF) {
        float* p = outF + (size_t)(b * PX + pix) * Cout + co0;
        if (beta) {
            v8f prev = *(const v8f*)p;
#pragma unroll
            for (int r = 0; r < 8; ++r) o[r] += prev[r];
        }
        *(v8f*)p = o;
    }
    if (outH) {
        v8h hv;
#pragma unroll
        for (int r = 0; r < 8; ++r) hv[r] = (_Float16)o[r];
        *(v8h*)(outH + ((size_t)b * PPX + padOff(pix)) * CtotH + c0H + co0) = hv;
    }
}

// ---------------------------------------------------------------------------
// Stride-2 encoder conv: x[:,t] (B,3,64,64) f32 -> padded NHWC (B,PPX,16) f16
// ---------------------------------------------------------------------------
__global__ void enc_conv_k(const float* __restrict__ x, int t,
                           const float* __restrict__ W,
                           const float* __restrict__ bias,
                           _Float16* __restrict__ outH)
{
    int i = blockIdx.x * blockDim.x + threadIdx.x;
    if (i >= BATCH * 16 * PX) return;
    int co = i & 15, pix = (i >> 4) & 1023, b = i >> 14;
    int y = pix >> 5, xo = pix & 31;
    float s = bias[co];
    for (int ci = 0; ci < 3; ++ci)
        for (int kh = 0; kh < 3; ++kh) {
            int yy = 2 * y + kh - 1;
            if ((unsigned)yy >= 64u) continue;
            for (int kw = 0; kw < 3; ++kw) {
                int xx = 2 * xo + kw - 1;
                if ((unsigned)xx >= 64u) continue;
                s += x[((((size_t)b * 8 + t) * 3 + ci) << 12) + (yy << 6) + xx]
                   * W[((co * 3 + ci) * 3 + kh) * 3 + kw];
            }
        }
    outH[((size_t)b * PPX + padOff(pix)) * 16 + co] = (_Float16)s;
}

// f32 dense NHWC (B,PX,C) -> f16 padded NHWC at channel offset c0; C=1<<cshift
__global__ void cvt_to_h_k(const float* __restrict__ src, _Float16* __restrict__ dst,
                           int cshift, int Ctot, int c0, int total)
{
    int i = blockIdx.x * blockDim.x + threadIdx.x;
    if (i >= total) return;
    int c = i & ((1 << cshift) - 1), bp = i >> cshift;
    int b = bp >> 10, pix = bp & 1023;
    dst[((size_t)b * PPX + padOff(pix)) * Ctot + c0 + c] = (_Float16)src[i];
}

// f16 padded NHWC (B,PPX,C) -> f16 padded NHWC at channel offset c0
__global__ void copy_h_k(const _Float16* __restrict__ src, _Float16* __restrict__ dst,
                         int cshift, int Ctot, int c0, int total)
{
    int i = blockIdx.x * blockDim.x + threadIdx.x;
    if (i >= total) return;
    int c = i & ((1 << cshift) - 1), bp = i >> cshift;
    int b = bp >> 10, pix = bp & 1023;
    size_t pp = (size_t)b * PPX + padOff(pix);
    dst[pp * Ctot + c0 + c] = src[pp * (1 << cshift) + c];
}

// r = sigmoid(g[:, :H]); write r*h (f16) into padded candCat at channel off 16
__global__ void make_rh_k(const float* __restrict__ g, const float* __restrict__ hs,
                          _Float16* __restrict__ dst, int hshift, int Ctot, int total)
{
    int i = blockIdx.x * blockDim.x + threadIdx.x;
    if (i >= total) return;
    int c = i & ((1 << hshift) - 1), bp = i >> hshift;
    int b = bp >> 10, pix = bp & 1023;
    float gv = g[((size_t)bp << (hshift + 1)) + c];
    float r  = 1.f / (1.f + __expf(-gv));
    dst[((size_t)b * PPX + padOff(pix)) * Ctot + 16 + c] = (_Float16)(r * hs[i]);
}

// h = (1-u)*h + u*tanh(cand), u = sigmoid(g[:, H:2H]); hs/g/cand dense NHWC,
// hsH padded NHWC
__global__ void gru_update_k(const float* __restrict__ g, const float* __restrict__ cand,
                             float* __restrict__ hs, _Float16* __restrict__ hsH,
                             int hshift, int total)
{
    int i = blockIdx.x * blockDim.x + threadIdx.x;
    if (i >= total) return;
    int c = i & ((1 << hshift) - 1), bp = i >> hshift;
    int b = bp >> 10, pix = bp & 1023;
    float gv = g[((size_t)bp << (hshift + 1)) + (1 << hshift) + c];
    float u  = 1.f / (1.f + __expf(-gv));
    float cn = tanhf(cand[i]);
    float h  = (1.f - u) * hs[i] + u * cn;
    hs[i] = h;
    hsH[((size_t)b * PPX + padOff(pix)) * (1 << hshift) + c] = (_Float16)h;
}

// p[b,o] = relu( sum_i relu(f[b,i]) * W[o,i] + bias[o] ), i is NCHW-flat;
// hs3 is dense NHWC (B,PX,64): f[b, c*1024+pix] = hs3[b, pix*64 + c]
__global__ void fc1_k(const float* __restrict__ hs3, const float* __restrict__ W,
                      const float* __restrict__ bias, float* __restrict__ p)
{
    int o = blockIdx.x, b = blockIdx.y;
    __shared__ float red[256];
    const float* f = hs3 + ((size_t)b << 16);
    const float* w = W   + ((size_t)o << 16);
    float s = 0.f;
    for (int i = threadIdx.x; i < 65536; i += 256) {
        int c = i >> 10, pix = i & 1023;
        s += fmaxf(f[(pix << 6) + c], 0.f) * w[i];
    }
    red[threadIdx.x] = s; __syncthreads();
    for (int st = 128; st > 0; st >>= 1) {
        if ((int)threadIdx.x < st) red[threadIdx.x] += red[threadIdx.x + st];
        __syncthreads();
    }
    if (threadIdx.x == 0) p[b * 100 + o] = fmaxf(red[0] + bias[o], 0.f);
}

__global__ void fc2_k(const float* __restrict__ p, const float* __restrict__ W,
                      const float* __restrict__ bias, float* __restrict__ out)
{
    int tid = blockIdx.x * blockDim.x + threadIdx.x;
    if (tid >= BATCH * 10) return;
    int b = tid / 10, o = tid - b * 10;
    float s = bias[o];
    for (int i = 0; i < 100; ++i) s += p[b * 100 + i] * W[o * 100 + i];
    out[tid] = s;
}

// ---------------------------------------------------------------------------
// Host orchestration
// ---------------------------------------------------------------------------
enum {
    PK_G0, PK_C0, PK_G1, PK_C1, PK_G2, PK_C2, PK_G3, PK_C3,
    PK_BUI0, PK_TD0_1, PK_TD0_2, PK_TDI0,
    PK_BU1_0, PK_BUI1, PK_TD1_2, PK_TDI1,
    PK_BU2_0, PK_BU2_1, PK_BUI2, PK_TD2_3, PK_TDI2,
    PK_BU3_2, PK_BUI3, PK_COUNT
};
static const int pkW[PK_COUNT]    = {1,3,5,7,9,11,13,15, 19,21,23,25, 27,29,31,33, 35,37,39,41,43, 45,47};
static const int pkCout[PK_COUNT] = {64,32,64,32,128,64,128,64, 16,16,16,16, 16,16,16,16, 16,16,16,16,16, 16,16};
static const int pkCin[PK_COUNT]  = {48,48,48,48,80,80,80,80, 16,32,64,32, 32,16,64,16, 32,32,32,64,16, 64,16};
static const int pkTr[PK_COUNT]   = {0,0,0,0,0,0,0,0, 0,1,1,0, 0,0,1,0, 0,0,0,1,0, 0,0};

static const int gPid[4]   = {PK_G0, PK_G1, PK_G2, PK_G3};
static const int cPid[4]   = {PK_C0, PK_C1, PK_C2, PK_C3};
static const int buIPid[4] = {PK_BUI0, PK_BUI1, PK_BUI2, PK_BUI3};
static const int tdIPid[4] = {PK_TDI0, PK_TDI1, PK_TDI2, -1};
static const int NIN[4]    = {0, 1, 2, 1};
static const int INJ[4][2] = {{-1,-1},{0,-1},{0,1},{2,-1}};
static const int buPid[4][2] = {{-1,-1},{PK_BU1_0,-1},{PK_BU2_0,PK_BU2_1},{PK_BU3_2,-1}};
static const int NOUT[4]   = {2, 1, 1, 0};
static const int OUTJ[4][2] = {{1,2},{2,-1},{3,-1},{-1,-1}};
static const int tdPid[4][2] = {{PK_TD0_1,PK_TD0_2},{PK_TD1_2,-1},{PK_TD2_3,-1},{-1,-1}};
static const int HIDs[4]   = {32, 32, 64, 64};
static const int HSH[4]    = {5, 5, 6, 6};   // log2(HID)
static const int ACT[4]    = {0, 1, 1, 2};

static inline size_t packElems(int Cout, int Cin) {
    return (size_t)((Cout + 15) / 16) * (size_t)((Cin * 9 + 31) / 32) * 512;
}

extern "C" void kernel_launch(void* const* d_in, const int* in_sizes, int n_in,
                              void* d_out, int out_size, void* d_ws, size_t ws_size,
                              hipStream_t stream)
{
    (void)in_sizes; (void)n_in; (void)out_size; (void)ws_size;
    const float* P[64];
    for (int i = 0; i < 53; ++i) P[i] = (const float*)d_in[i];
    const float* x = P[0];

    // ---- workspace layout: f16 conv inputs halo-padded NHWC, f32 dense NHWC
    char* ws = (char*)d_ws;
    size_t off = 0;
    auto alloc = [&](size_t bytes) -> void* {
        void* p = ws + off; off = (off + bytes + 255) & ~(size_t)255; return p;
    };
    float*     hs[4];
    _Float16*  hsH[4];
    for (int n = 0; n < 4; ++n) {
        hs[n]  = (float*)    alloc((size_t)BATCH * PX  * HIDs[n] * 4);
        hsH[n] = (_Float16*) alloc((size_t)BATCH * PPX * HIDs[n] * 2);
    }
    size_t wpTotal = 0;
    size_t wpOff[PK_COUNT];
    for (int p = 0; p < PK_COUNT; ++p) { wpOff[p] = wpTotal; wpTotal += packElems(pkCout[p], pkCin[p]); }
    _Float16* Wpool   = (_Float16*)alloc(wpTotal * 2);
    const size_t zBu = (size_t)BATCH * PPX * 32 * 2;
    const size_t zCat = (size_t)BATCH * PPX * 80 * 2;
    _Float16* buCat   = (_Float16*)alloc(zBu);
    _Float16* tdCat   = (_Float16*)alloc(zBu);
    _Float16* gateCat = (_Float16*)alloc(zCat);
    _Float16* candCat = (_Float16*)alloc(zCat);
    float* inpF = (float*)alloc((size_t)BATCH * PX * 16  * 4);
    float* gBuf = (float*)alloc((size_t)BATCH * PX * 128 * 4);
    float* cBuf = (float*)alloc((size_t)BATCH * PX * 64  * 4);
    float* pBuf = (float*)alloc((size_t)BATCH * 100 * 4);

    // ---- zero hidden state + halos of all padded f16 buffers (interiors are
    // always fully overwritten before being read; halos must stay zero) ----
    for (int n = 0; n < 4; ++n) {
        hipMemsetAsync(hs[n],  0, (size_t)BATCH * PX  * HIDs[n] * 4, stream);
        hipMemsetAsync(hsH[n], 0, (size_t)BATCH * PPX * HIDs[n] * 2, stream);
    }
    hipMemsetAsync(buCat,   0, zBu,  stream);
    hipMemsetAsync(tdCat,   0, zBu,  stream);
    hipMemsetAsync(gateCat, 0, zCat, stream);
    hipMemsetAsync(candCat, 0, zCat, stream);

    // ---- pack all conv weights into WMMA A-fragment layout (tap-major K) ----
    for (int p = 0; p < PK_COUNT; ++p) {
        int total = (int)packElems(pkCout[p], pkCin[p]);
        pack_w_k<<<(total + 255) / 256, 256, 0, stream>>>(
            P[pkW[p]], Wpool + wpOff[p], pkCout[p], pkCin[p], pkTr[p], total);
    }

    auto conv = [&](const _Float16* in, int Cin, int pid,
                    float* outF, int beta, _Float16* outH, int Ctot, int c0) {
        dim3 g(pkCout[pid] / 16, 8, BATCH);
        const _Float16* wp = Wpool + wpOff[pid];
        const float* bs = P[pkW[pid] + 1];
        int Cout = pkCout[pid];
        switch (Cin) {
        case 16: conv3x3_wmma<16><<<g,256,0,stream>>>(in, wp, bs, Cout, outF, beta, outH, Ctot, c0); break;
        case 32: conv3x3_wmma<32><<<g,256,0,stream>>>(in, wp, bs, Cout, outF, beta, outH, Ctot, c0); break;
        case 48: conv3x3_wmma<48><<<g,256,0,stream>>>(in, wp, bs, Cout, outF, beta, outH, Ctot, c0); break;
        case 64: conv3x3_wmma<64><<<g,256,0,stream>>>(in, wp, bs, Cout, outF, beta, outH, Ctot, c0); break;
        case 80: conv3x3_wmma<80><<<g,256,0,stream>>>(in, wp, bs, Cout, outF, beta, outH, Ctot, c0); break;
        }
    };
    auto blocks = [](int total) { return dim3((total + 255) / 256); };

    // ---- recurrent graph: t-major, node order preserves hs aliasing ----
    for (int t = 0; t < 11; ++t) {
        for (int n = 0; n < 4; ++n) {
            if (t < ACT[n]) continue;
            if (n == 0 && t >= 8) continue;   // bu list empty for node 0
            const int H = HIDs[n], hsh = HSH[n];
            const int numin = NIN[n] + (n == 0 ? 1 : 0);
            int cpos = 0;
            // --- bottom-up features into buCat (f16 padded NHWC concat) ---
            if (n == 0) {
                enc_conv_k<<<blocks(BATCH * 16 * PX), 256, 0, stream>>>(
                    x, t, P[17], P[18], buCat);
                cpos = 1;
            }
            for (int s = 0; s < NIN[n]; ++s) {
                int j = INJ[n][s];
                conv(hsH[j], HIDs[j], buPid[n][s],
                     nullptr, 0, buCat, 16 * numin, 16 * cpos);
                ++cpos;
            }
            // --- integrator conv -> inp (f32 dense NHWC) ---
            conv(buCat, 16 * numin, buIPid[n], inpF, 0, nullptr, 0, 0);
            // --- top-down (transpose convs) accumulate into inp ---
            if (t != 0 && NOUT[n] > 0) {
                for (int s = 0; s < NOUT[n]; ++s) {
                    int j = OUTJ[n][s];
                    conv(hsH[j], HIDs[j], tdPid[n][s],
                         nullptr, 0, tdCat, 16 * NOUT[n], 16 * s);
                }
                conv(tdCat, 16 * NOUT[n], tdIPid[n], inpF, 1, nullptr, 0, 0);
            }
            // --- GRU cell ---
            const int Ctot = 16 + H;
            const int tI = BATCH * PX * 16;
            const int tH = BATCH * PX * H;
            cvt_to_h_k<<<blocks(tI), 256, 0, stream>>>(inpF, gateCat, 4, Ctot, 0, tI);
            cvt_to_h_k<<<blocks(tI), 256, 0, stream>>>(inpF, candCat, 4, Ctot, 0, tI);
            copy_h_k  <<<blocks(tH), 256, 0, stream>>>(hsH[n], gateCat, hsh, Ctot, 16, tH);
            conv(gateCat, Ctot, gPid[n], gBuf, 0, nullptr, 0, 0);
            make_rh_k <<<blocks(tH), 256, 0, stream>>>(gBuf, hs[n], candCat, hsh, Ctot, tH);
            conv(candCat, Ctot, cPid[n], cBuf, 0, nullptr, 0, 0);
            gru_update_k<<<blocks(tH), 256, 0, stream>>>(gBuf, cBuf, hs[n], hsH[n], hsh, tH);
        }
    }

    // ---- head: relu -> fc1 (relu) -> fc2 ----
    fc1_k<<<dim3(100, BATCH), 256, 0, stream>>>(hs[3], P[49], P[50], pBuf);
    fc2_k<<<1, 320, 0, stream>>>(pBuf, P[51], P[52], (float*)d_out);
}
